// EncoderBlock_3891240370896
// MI455X (gfx1250) — compile-verified, matrix operand order checked by reading, not run on hardware
//
#include <hip/hip_runtime.h>
#include <hip/hip_bf16.h>

// ---------------------------------------------------------------------------
// Fused encoder block for MI455X (gfx1250, wave32, WMMA).
// One workgroup (256 threads = 8 waves) per batch element. All activations
// stay resident in LDS; every matmul is V_WMMA_F32_16X16X32_F16.
// ---------------------------------------------------------------------------

typedef _Float16 v8h  __attribute__((ext_vector_type(8)));
typedef _Float16 v16h __attribute__((ext_vector_type(16)));
typedef float    v8f  __attribute__((ext_vector_type(8)));

#define DD      128
#define LL      128
#define ACT_LD  129     // fp32 tile row stride (bank-staggered)
#define H_LD    136     // fp16 staging row stride (272B = 17*16B, keeps 16B align)
#define NTHREADS 256

// Load one 16x32 fp16 fragment (A or B operand) from an LDS plane laid out
// row-major with row stride H_LD. Per ISA 7.12.2 (16-bit A 16x32):
//   lanes 0-15 : row = row0+lane,    halves 0..7 = K k0..k0+7,  8..15 = K k0+16..k0+23
//   lanes16-31 : row = row0+lane-16, halves 0..7 = K k0+8..k0+15, 8..15 = K k0+24..k0+31
__device__ __forceinline__ v16h load_frag(const _Float16* base, int row0, int k0, int lane) {
  int r  = row0 + (lane & 15);
  int kk = k0 + ((lane >> 4) << 3);          // +0 or +8
  const _Float16* p = base + r * H_LD + kk;  // 16B aligned
  v8h lo = *(const v8h*)(p);
  v8h hi = *(const v8h*)(p + 16);
  v16h out;
#pragma unroll
  for (int i = 0; i < 8; ++i) { out[i] = lo[i]; out[i + 8] = hi[i]; }
  return out;
}

// One 16x16 output tile of C = A * B^T  (A: [M][K], B: [N][K], both fp16 in LDS)
__device__ __forceinline__ v8f gemm_tile(const _Float16* A, const _Float16* B,
                                         int m0, int n0, int kStart, int kSteps,
                                         int lane) {
  v8f acc = {0.f, 0.f, 0.f, 0.f, 0.f, 0.f, 0.f, 0.f};
  for (int ks = 0; ks < kSteps; ++ks) {
    int k0 = kStart + ks * 32;
    v16h a = load_frag(A, m0, k0, lane);
    v16h b = load_frag(B, n0, k0, lane);
    acc = __builtin_amdgcn_wmma_f32_16x16x32_f16(false, a, false, b,
                                                 (short)0, acc, false, false);
  }
  return acc;
}

// C-fragment coordinates: element j of the v8f for this lane sits at
//   row = m0 + ((lane>>4)<<3) + j , col = n0 + (lane&15)
#define C_COL(n0, lane)  ((n0) + ((lane) & 15))
#define C_ROWB(m0, lane) ((m0) + (((lane) >> 4) << 3))

__device__ __forceinline__ void layernorm2d(float* act, const float* __restrict__ gw,
                                            const float* __restrict__ gb,
                                            float* redS, float* redQ, int tid) {
  __syncthreads();                 // protect redS/redQ reuse
  float s = 0.f, q = 0.f;
#pragma unroll 4
  for (int i = 0; i < 64; ++i) {
    int idx = tid + i * NTHREADS;
    int d = idx >> 7, l = idx & 127;
    float v = act[d * ACT_LD + l];
    s += v; q += v * v;
  }
  redS[tid] = s; redQ[tid] = q;
  __syncthreads();
  for (int stp = 128; stp > 0; stp >>= 1) {
    if (tid < stp) { redS[tid] += redS[tid + stp]; redQ[tid] += redQ[tid + stp]; }
    __syncthreads();
  }
  float mu   = redS[0] * (1.f / 16384.f);
  float var  = redQ[0] * (1.f / 16384.f) - mu * mu;
  float rstd = rsqrtf(var + 1e-5f);
#pragma unroll 4
  for (int i = 0; i < 64; ++i) {
    int idx = tid + i * NTHREADS;
    int d = idx >> 7, l = idx & 127;
    float v = act[d * ACT_LD + l];
    act[d * ACT_LD + l] = (v - mu) * rstd * gw[idx] + gb[idx];
  }
  __syncthreads();
}

__global__ __launch_bounds__(NTHREADS) void encoder_block_kernel(
    const float* __restrict__ x,    const int* __restrict__ mask,
    const float* __restrict__ dw_w, const float* __restrict__ dw_b,
    const float* __restrict__ pw_w, const float* __restrict__ pw_b,
    const float* __restrict__ normb_w, const float* __restrict__ normb_b,
    const float* __restrict__ norms_w, const float* __restrict__ norms_b,
    const float* __restrict__ norme_w, const float* __restrict__ norme_b,
    const float* __restrict__ Wq, const float* __restrict__ bq,
    const float* __restrict__ Wk, const float* __restrict__ bk,
    const float* __restrict__ Wv, const float* __restrict__ bv,
    const float* __restrict__ Wo, const float* __restrict__ bo,
    const float* __restrict__ fc_w, const float* __restrict__ fc_b,
    float* __restrict__ out) {

  __shared__ __attribute__((aligned(16))) float    s_act[DD * ACT_LD]; // activation / score scratch
  __shared__ __attribute__((aligned(16))) float    s_res[DD * ACT_LD]; // residual
  __shared__ __attribute__((aligned(16))) _Float16 s_A [DD * H_LD];    // GEMM A staging
  __shared__ __attribute__((aligned(16))) _Float16 s_B [DD * H_LD];    // GEMM B staging / O matrix
  __shared__ __attribute__((aligned(16))) _Float16 s_Q [DD * H_LD];    // Q (fp16) / Wo staging
  __shared__ __attribute__((aligned(16))) _Float16 s_K [DD * H_LD];    // K (fp16)
  __shared__ __attribute__((aligned(16))) _Float16 s_Vt[DD * H_LD];    // V transposed [f][s]
  __shared__ float s_cw[DD * 8];      // depthwise weights(7)+bias(1) per channel
  __shared__ float s_mask[LL];
  __shared__ float s_redS[NTHREADS], s_redQ[NTHREADS];

  const int b    = blockIdx.x;
  const int tid  = threadIdx.x;
  const int lane = tid & 31;
  const int wave = tid >> 5;
  const float* xb = x + (size_t)b * (DD * LL);

  if (tid < LL) s_mask[tid] = (float)mask[(size_t)b * LL + tid];

  // ---- x + positional encoding; residual = result -------------------------
  const float KLOG = 9.2103403719761836f / 128.f;   // ln(10000)/D
#pragma unroll 2
  for (int i = 0; i < 64; ++i) {
    int idx = tid + i * NTHREADS;
    int d = idx >> 7, l = idx & 127;
    float dd = (float)(d & ~1);
    float f  = __expf(-dd * KLOG);
    float ang = (float)l * f;
    float pe = (d & 1) ? __cosf(ang) : __sinf(ang);
    float v  = xb[idx] + pe;
    s_act[d * ACT_LD + l] = v;
    s_res[d * ACT_LD + l] = v;
  }
  __syncthreads();
  layernorm2d(s_act, normb_w, normb_b, s_redS, s_redQ, tid);

  // ---- 4 separable-conv layers --------------------------------------------
  for (int layer = 0; layer < 4; ++layer) {
    // depthwise weights + bias -> LDS ; pointwise weights -> fp16 A plane
    for (int i = tid; i < DD * 8; i += NTHREADS) {
      int c = i >> 3, j = i & 7;
      s_cw[i] = (j < 7) ? dw_w[(layer * DD + c) * 7 + j] : dw_b[layer * DD + c];
    }
#pragma unroll 2
    for (int i = 0; i < 64; ++i) {
      int idx = tid + i * NTHREADS;
      int o = idx >> 7, c = idx & 127;
      s_A[o * H_LD + c] = (_Float16)pw_w[layer * DD * DD + idx];
    }
    __syncthreads();

    // depthwise conv -> s_B transposed: s_B[l][c] = h[c][l]  (B operand [N][K])
#pragma unroll 2
    for (int i = 0; i < 64; ++i) {
      int idx = tid + i * NTHREADS;
      int c = idx >> 7, l = idx & 127;
      float acc = s_cw[c * 8 + 7];
#pragma unroll
      for (int j = 0; j < 7; ++j) {
        int ll = l + j - 3;
        if (ll >= 0 && ll < LL) acc += s_cw[c * 8 + j] * s_act[c * ACT_LD + ll];
      }
      s_B[l * H_LD + c] = (_Float16)acc;
    }
    __syncthreads();

    // pointwise GEMM: out[o][l] = pw_w[o][:] . h[:][l] ; +bias, ReLU, +res
    for (int t = 0; t < 8; ++t) {
      int tile = wave * 8 + t;
      int m0 = (tile >> 3) << 4, n0 = (tile & 7) << 4;
      v8f c = gemm_tile(s_A, s_B, m0, n0, 0, 4, lane);
      int n = C_COL(n0, lane), mb = C_ROWB(m0, lane);
#pragma unroll
      for (int j = 0; j < 8; ++j) {
        int m = mb + j;
        float v = c[j] + pw_b[layer * DD + m];
        v = fmaxf(v, 0.f) + s_res[m * ACT_LD + n];
        s_act[m * ACT_LD + n] = v;
        s_res[m * ACT_LD + n] = v;
      }
    }
    __syncthreads();
    layernorm2d(s_act, norms_w + layer * DD * LL, norms_b + layer * DD * LL,
                s_redS, s_redQ, tid);
  }

  // ---- MHA: rows of the (D,L) tile are the sequence, cols the features ----
  // Stage X (fp16) as the shared A operand for Q/K/V projections.
#pragma unroll 2
  for (int i = 0; i < 64; ++i) {
    int idx = tid + i * NTHREADS;
    int d = idx >> 7, l = idx & 127;
    s_A[d * H_LD + l] = (_Float16)s_act[d * ACT_LD + l];
  }

  // --- Q / K / V projections (B operand = weight row-major = [N][K]) ------
  for (int proj = 0; proj < 3; ++proj) {
    const float* W  = (proj == 0) ? Wq : (proj == 1) ? Wk : Wv;
    const float* bb = (proj == 0) ? bq : (proj == 1) ? bk : bv;
    __syncthreads();                 // prior GEMM done before restaging s_B
#pragma unroll 2
    for (int i = 0; i < 64; ++i) {
      int idx = tid + i * NTHREADS;
      int f = idx >> 7, l = idx & 127;
      s_B[f * H_LD + l] = (_Float16)W[idx];
    }
    __syncthreads();
    for (int t = 0; t < 8; ++t) {
      int tile = wave * 8 + t;
      int m0 = (tile >> 3) << 4, n0 = (tile & 7) << 4;
      v8f c = gemm_tile(s_A, s_B, m0, n0, 0, 4, lane);
      int n = C_COL(n0, lane), mb = C_ROWB(m0, lane);
#pragma unroll
      for (int j = 0; j < 8; ++j) {
        int m = mb + j;
        float v = c[j] + bb[n];
        if (proj == 0)      s_Q[m * H_LD + n]  = (_Float16)v;
        else if (proj == 1) s_K[m * H_LD + n]  = (_Float16)v;
        else                s_Vt[n * H_LD + m] = (_Float16)v;  // transposed
      }
    }
  }
  __syncthreads();

  // --- per-head attention; O accumulated (fp16) into s_B -------------------
  const float scale = 0.17677669529663687f;   // 1/sqrt(32)
  for (int h = 0; h < 4; ++h) {
    // scores[q][k] (k over sequence rows), masked+scaled, into s_act scratch
    for (int t = 0; t < 8; ++t) {
      int tile = wave * 8 + t;
      int m0 = (tile >> 3) << 4, n0 = (tile & 7) << 4;
      v8f c = gemm_tile(s_Q, s_K, m0, n0, h * 32, 1, lane);
      int n = C_COL(n0, lane), mb = C_ROWB(m0, lane);
      float mv = s_mask[n];
#pragma unroll
      for (int j = 0; j < 8; ++j) {
        float sv = c[j] * scale;
        s_act[(mb + j) * ACT_LD + n] = sv * (1.f - mv) + mv * (-1e30f);
      }
    }
    __syncthreads();

    // softmax per row -> attn (fp16) in s_A
    if (tid < 128) {
      float mx = -3.4e38f;
      for (int k = 0; k < LL; ++k) mx = fmaxf(mx, s_act[tid * ACT_LD + k]);
      float sum = 0.f;
      for (int k = 0; k < LL; ++k) {
        float e = __expf(s_act[tid * ACT_LD + k] - mx);
        s_act[tid * ACT_LD + k] = e;
        sum += e;
      }
      float inv = 1.f / sum;
      for (int k = 0; k < LL; ++k)
        s_A[tid * H_LD + k] = (_Float16)(s_act[tid * ACT_LD + k] * inv);
    }
    __syncthreads();

    // O[:, h*32 : h*32+32] = attn @ V_h   (B operand = s_Vt rows = features)
    for (int t = 0; t < 2; ++t) {
      int tile = wave * 2 + t;
      int m0 = (tile >> 1) << 4;
      int nf = h * 32 + ((tile & 1) << 4);
      v8f c = gemm_tile(s_A, s_Vt, m0, nf, 0, 4, lane);
      int n = C_COL(nf, lane), mb = C_ROWB(m0, lane);
#pragma unroll
      for (int j = 0; j < 8; ++j) s_B[(mb + j) * H_LD + n] = (_Float16)c[j];
    }
    __syncthreads();
  }

  // --- output projection: att = O @ Wo^T + bo ; out = att + res ; LN -------
#pragma unroll 2
  for (int i = 0; i < 64; ++i) {
    int idx = tid + i * NTHREADS;
    int g = idx >> 7, f = idx & 127;
    s_Q[g * H_LD + f] = (_Float16)Wo[idx];   // reuse Q plane for Wo
  }
  __syncthreads();
  for (int t = 0; t < 8; ++t) {
    int tile = wave * 8 + t;
    int m0 = (tile >> 3) << 4, n0 = (tile & 7) << 4;
    v8f c = gemm_tile(s_B, s_Q, m0, n0, 0, 4, lane);
    int n = C_COL(n0, lane), mb = C_ROWB(m0, lane);
#pragma unroll
    for (int j = 0; j < 8; ++j) {
      int m = mb + j;
      float v = c[j] + bo[n] + s_res[m * ACT_LD + n];
      s_act[m * ACT_LD + n] = v;
      s_res[m * ACT_LD + n] = v;
    }
  }
  __syncthreads();
  layernorm2d(s_act, norme_w, norme_b, s_redS, s_redQ, tid);

  // --- final FC: out[o][l] = relu(fc_w[o][:] . act[:][l] + fc_b[o]) + res --
#pragma unroll 2
  for (int i = 0; i < 64; ++i) {
    int idx = tid + i * NTHREADS;
    int o = idx >> 7, c2 = idx & 127;
    s_A[o * H_LD + c2] = (_Float16)fc_w[idx];
  }
#pragma unroll 2
  for (int i = 0; i < 64; ++i) {
    int idx = tid + i * NTHREADS;
    int c2 = idx >> 7, l = idx & 127;
    s_B[l * H_LD + c2] = (_Float16)s_act[c2 * ACT_LD + l];   // act^T for B operand
  }
  __syncthreads();

  float* outb = out + (size_t)b * (DD * LL);
  for (int t = 0; t < 8; ++t) {
    int tile = wave * 8 + t;
    int m0 = (tile >> 3) << 4, n0 = (tile & 7) << 4;
    v8f c = gemm_tile(s_A, s_B, m0, n0, 0, 4, lane);
    int n = C_COL(n0, lane), mb = C_ROWB(m0, lane);
#pragma unroll
    for (int j = 0; j < 8; ++j) {
      int m = mb + j;
      float v = c[j] + fc_b[m];
      v = fmaxf(v, 0.f) + s_res[m * ACT_LD + n];
      outb[m * LL + n] = v;
    }
  }
}

extern "C" void kernel_launch(void* const* d_in, const int* in_sizes, int n_in,
                              void* d_out, int out_size, void* d_ws, size_t ws_size,
                              hipStream_t stream) {
  const float* x       = (const float*)d_in[0];
  const int*   mask    = (const int*)  d_in[1];
  const float* dw_w    = (const float*)d_in[2];
  const float* dw_b    = (const float*)d_in[3];
  const float* pw_w    = (const float*)d_in[4];
  const float* pw_b    = (const float*)d_in[5];
  const float* normb_w = (const float*)d_in[6];
  const float* normb_b = (const float*)d_in[7];
  const float* norms_w = (const float*)d_in[8];
  const float* norms_b = (const float*)d_in[9];
  const float* norme_w = (const float*)d_in[10];
  const float* norme_b = (const float*)d_in[11];
  const float* Wq      = (const float*)d_in[12];
  const float* bq      = (const float*)d_in[13];
  const float* Wk      = (const float*)d_in[14];
  const float* bk      = (const float*)d_in[15];
  const float* Wv      = (const float*)d_in[16];
  const float* bv      = (const float*)d_in[17];
  const float* Wo      = (const float*)d_in[18];
  const float* bo      = (const float*)d_in[19];
  const float* fc_w    = (const float*)d_in[20];
  const float* fc_b    = (const float*)d_in[21];

  int B = in_sizes[0] / (DD * LL);
  encoder_block_kernel<<<B, NTHREADS, 0, stream>>>(
      x, mask, dw_w, dw_b, pw_w, pw_b, normb_w, normb_b, norms_w, norms_b,
      norme_w, norme_b, Wq, bq, Wk, bk, Wv, bv, Wo, bo, fc_w, fc_b,
      (float*)d_out);
}
